// ConditionalDenoiseEncoderTransformer_63625645523437
// MI455X (gfx1250) — matrix operand order repaired and checked by reading, not compiled, not run on hardware
//
#include <hip/hip_runtime.h>
#include <hip/hip_bf16.h>
#include <stdint.h>

#define N_TOK   1024
#define N_IMG   512
#define N_TXT   512
#define VOCAB_SZ 10
#define DMODEL  1024
#define NLAYER  12
#define HMLP    4096
#define BATCH   8
#define LN_EPS  1e-5f

typedef __attribute__((ext_vector_type(16))) __bf16 bf16x16;
typedef __attribute__((ext_vector_type(8)))  float  f32x8;
typedef __attribute__((ext_vector_type(4)))  unsigned int u32x4;

// ---------- helpers ----------
__device__ __forceinline__ unsigned short f2bf(float f) {
  union { float f; unsigned int u; } c; c.f = f;
  unsigned int u = c.u;
  u += 0x7fffu + ((u >> 16) & 1u);   // round-to-nearest-even
  return (unsigned short)(u >> 16);
}

__device__ __forceinline__ float block_sum(float v) {
  __shared__ float sb[8];
  for (int o = 16; o > 0; o >>= 1) v += __shfl_down(v, o, 32);
  if ((threadIdx.x & 31) == 0) sb[threadIdx.x >> 5] = v;
  __syncthreads();
  float t = 0.f;
#pragma unroll
  for (int i = 0; i < 8; i++) t += sb[i];
  __syncthreads();
  return t;
}

__device__ __forceinline__ float block_max(float v) {
  __shared__ float sb[8];
  for (int o = 16; o > 0; o >>= 1) v = fmaxf(v, __shfl_down(v, o, 32));
  if ((threadIdx.x & 31) == 0) sb[threadIdx.x >> 5] = v;
  __syncthreads();
  float t = sb[0];
#pragma unroll
  for (int i = 1; i < 8; i++) t = fmaxf(t, sb[i]);
  __syncthreads();
  return t;
}

// ---------- WMMA fragments ----------
union FragU { bf16x16 v; u32x4 u[2]; };

// A 16x32 bf16 tile, row-major source (ldk = row stride in elements).
// lanes 0-15: row m=lane, K 0-7 (u[0]) and K 16-23 (u[1])
// lanes 16-31: row m=lane-16, K 8-15 (u[0]) and K 24-31 (u[1])
__device__ __forceinline__ FragU load_fragA(const unsigned short* base, int ldk, int lane) {
  FragU f;
  const int r  = lane & 15;
  const int kh = (lane >> 4) & 1;
  const unsigned short* p = base + (long long)r * ldk + kh * 8;
  f.u[0] = *reinterpret_cast<const u32x4*>(p);
  f.u[1] = *reinterpret_cast<const u32x4*>(p + 16);
  return f;
}

// B 32x16 bf16 tile from Bt[n][k] row-major (column-major B).
// lanes 0-15: col n=lane, K 0-7 (u[0]) and K 16-23 (u[1]); lanes 16-31: col n=lane-16, K 8-15 (u[0]) and K 24-31 (u[1])
__device__ __forceinline__ FragU load_fragB(const unsigned short* base, int ldk, int lane) {
  FragU f;
  const int n  = lane & 15;
  const int kh = (lane >> 4) & 1;
  const unsigned short* p = base + (long long)n * ldk + kh * 8;
  f.u[0] = *reinterpret_cast<const u32x4*>(p);
  f.u[1] = *reinterpret_cast<const u32x4*>(p + 16);
  return f;
}

// ---------- GEMM: C[M,Nn] = A[M,K] @ Bt[Nn,K]^T  (+ epilogue) ----------
// Wave tile 64x64 (16 WMMAs / 16 b128 loads per K-step of 32).
// Block: 8 waves as 2 (M) x 4 (N) -> 128 x 256 block tile.
// MODE 0: store bf16
// MODE 1: store f32 * scale           (attention scores)
// MODE 2: f32 out += c                (attn @ v residual add)
// MODE 3: bf16 out = gelu(c + bias)   (MLP up)
// MODE 4: f32 out += c + bias         (MLP down residual add)
template <int MODE>
__global__ __launch_bounds__(256)
void gemm_bf16_wmma(const unsigned short* __restrict__ A,
                    const unsigned short* __restrict__ Bt,
                    void* __restrict__ OutP,
                    const float* __restrict__ bias,
                    int M, int Nn, int K,
                    long long aBatch, long long bBatch, long long oBatch,
                    float scale)
{
  const int lane = threadIdx.x & 31;
  const int w    = threadIdx.x >> 5;
  const int wm   = w & 1;          // 2 waves along M
  const int wn   = w >> 1;         // 4 waves along N
  const long long m0 = (long long)blockIdx.y * 128 + wm * 64;
  const long long n0 = (long long)blockIdx.x * 256 + wn * 64;
  const unsigned short* Ab = A  + (long long)blockIdx.z * aBatch;
  const unsigned short* Bb = Bt + (long long)blockIdx.z * bBatch;

  f32x8 acc[4][4];
#pragma unroll
  for (int i = 0; i < 4; i++)
#pragma unroll
    for (int j = 0; j < 4; j++) acc[i][j] = {};

  const int ar = lane & 15;

  for (int k0 = 0; k0 < K; k0 += 32) {
    FragU a[4], b[4];
#pragma unroll
    for (int i = 0; i < 4; i++)
      a[i] = load_fragA(Ab + (m0 + i * 16) * K + k0, K, lane);
#pragma unroll
    for (int j = 0; j < 4; j++)
      b[j] = load_fragB(Bb + (n0 + j * 16) * K + k0, K, lane);

    if (k0 + 32 < K) {  // keep the L2 -> WGP$ stream ahead of the WMMA pipe
      __builtin_prefetch(Ab + (m0 + ar) * K + k0 + 32, 0, 1);
      __builtin_prefetch(Bb + (n0 + ar) * K + k0 + 32, 0, 1);
    }

#pragma unroll
    for (int i = 0; i < 4; i++)
#pragma unroll
      for (int j = 0; j < 4; j++)
        acc[i][j] = __builtin_amdgcn_wmma_f32_16x16x32_bf16(
            false, a[i].v, false, b[j].v, (short)0, acc[i][j], false, false);
  }

  const int col = lane & 15;
  const int ro  = (lane >> 4) << 3;   // +8 rows for upper lane half

#pragma unroll
  for (int i = 0; i < 4; i++) {
#pragma unroll
    for (int j = 0; j < 4; j++) {
      const long long mb = m0 + i * 16 + ro;
      const long long n  = n0 + j * 16 + col;
      float bv = 0.f;
      if constexpr (MODE == 3 || MODE == 4) bv = bias[n];
      f32x8 c = acc[i][j];
#pragma unroll
      for (int r = 0; r < 8; r++) {
        const long long idx = (long long)blockIdx.z * oBatch + (mb + r) * (long long)Nn + n;
        float x = c[r];
        if constexpr (MODE == 0) {
          ((unsigned short*)OutP)[idx] = f2bf(x);
        } else if constexpr (MODE == 1) {
          ((float*)OutP)[idx] = x * scale;
        } else if constexpr (MODE == 2) {
          ((float*)OutP)[idx] += x;
        } else if constexpr (MODE == 3) {
          float t = x + bv;
          float g = 0.5f * t * (1.0f + erff(t * 0.70710678118654752f));
          ((unsigned short*)OutP)[idx] = f2bf(g);
        } else {
          ((float*)OutP)[idx] += x + bv;
        }
      }
    }
  }
}

// ---------- embedding ----------
__global__ __launch_bounds__(256)
void embed_kernel(const int* __restrict__ xt, const int* __restrict__ zi,
                  const float* __restrict__ pos, const float* __restrict__ temb,
                  float* __restrict__ H)
{
  const long long idx = (long long)blockIdx.x * 256 + threadIdx.x;
  const int d = (int)(idx & (DMODEL - 1));
  const long long bt = idx >> 10;
  const int t = (int)(bt & (N_TOK - 1));
  const int b = (int)(bt >> 10);
  float e;
  if (t < N_IMG) {
    if (d < VOCAB_SZ) {
      float dd = (float)d - (float)zi[b * N_IMG + t];
      e = -0.5f * dd * dd;
    } else e = 0.f;
  } else {
    int tok = xt[b * N_TXT + (t - N_IMG)];
    e = temb[(long long)tok * DMODEL + d];
  }
  H[idx] = e + pos[(long long)t * DMODEL + d];
}

// ---------- layernorm (fp32 in -> bf16 out) ----------
__global__ __launch_bounds__(256)
void layernorm_bf16(const float* __restrict__ X, const float* __restrict__ g,
                    const float* __restrict__ b, unsigned short* __restrict__ Y)
{
  const long long row = blockIdx.x;
  const float* x = X + row * DMODEL;
  float s = 0.f, ss = 0.f;
  for (int i = threadIdx.x; i < DMODEL; i += 256) { float v = x[i]; s += v; ss += v * v; }
  s  = block_sum(s);
  ss = block_sum(ss);
  const float mu   = s * (1.0f / DMODEL);
  const float var  = ss * (1.0f / DMODEL) - mu * mu;
  const float rstd = rsqrtf(var + LN_EPS);
  unsigned short* y = Y + row * DMODEL;
  for (int i = threadIdx.x; i < DMODEL; i += 256)
    y[i] = f2bf((x[i] - mu) * rstd * g[i] + b[i]);
}

// ---------- softmax (fp32 in -> bf16 out), row length N_TOK ----------
__global__ __launch_bounds__(256)
void softmax_bf16(const float* __restrict__ S, unsigned short* __restrict__ P)
{
  __shared__ float e[N_TOK];
  const long long row = blockIdx.x;
  const float* s = S + row * N_TOK;
  float mx = -3.4e38f;
  for (int i = threadIdx.x; i < N_TOK; i += 256) mx = fmaxf(mx, s[i]);
  mx = block_max(mx);
  float sum = 0.f;
  for (int i = threadIdx.x; i < N_TOK; i += 256) { float v = expf(s[i] - mx); e[i] = v; sum += v; }
  sum = block_sum(sum);
  const float rinv = 1.0f / sum;
  unsigned short* p = P + row * N_TOK;
  for (int i = threadIdx.x; i < N_TOK; i += 256) p[i] = f2bf(e[i] * rinv);
}

// ---------- fp32 -> bf16 transposed weight staging (src[R][C] -> dst[C][R]) ----------
__global__ __launch_bounds__(256)
void transpose_cast_f32_bf16(const float* __restrict__ src, unsigned short* __restrict__ dst,
                             int R, int C)
{
  __shared__ float tile[32][33];
  const int tx = threadIdx.x & 31, ty = threadIdx.x >> 5;
  const int c0 = blockIdx.x * 32, r0 = blockIdx.y * 32;
#pragma unroll
  for (int j = 0; j < 32; j += 8)
    tile[ty + j][tx] = src[(long long)(r0 + ty + j) * C + (c0 + tx)];
  __syncthreads();
#pragma unroll
  for (int j = 0; j < 32; j += 8)
    dst[(long long)(c0 + ty + j) * R + (r0 + tx)] = f2bf(tile[tx][ty + j]);
}

// ---------- batched bf16 transpose (src[z][R][C] -> dst[z][C][R]) ----------
__global__ __launch_bounds__(256)
void transpose_bf16_batched(const unsigned short* __restrict__ src,
                            unsigned short* __restrict__ dst, int R, int C)
{
  __shared__ unsigned short tile[32][34];
  const int tx = threadIdx.x & 31, ty = threadIdx.x >> 5;
  const int c0 = blockIdx.x * 32, r0 = blockIdx.y * 32;
  const unsigned short* s = src + (long long)blockIdx.z * R * C;
  unsigned short* d = dst + (long long)blockIdx.z * R * C;
#pragma unroll
  for (int j = 0; j < 32; j += 8)
    tile[ty + j][tx] = s[(long long)(r0 + ty + j) * C + (c0 + tx)];
  __syncthreads();
#pragma unroll
  for (int j = 0; j < 32; j += 8)
    d[(long long)(c0 + ty + j) * R + (r0 + tx)] = tile[tx][ty + j];
}

// ---------- readout: out[b][t] = H[b][t] . ro_w + ro_b ----------
__global__ __launch_bounds__(256)
void readout_kernel(const float* __restrict__ H, const float* __restrict__ w,
                    const float* __restrict__ b, float* __restrict__ out)
{
  const int bi = blockIdx.x / N_IMG;
  const int t  = blockIdx.x % N_IMG;
  const float* h = H + ((long long)bi * N_TOK + t) * DMODEL;
  float s = 0.f;
  for (int i = threadIdx.x; i < DMODEL; i += 256) s += h[i] * w[i];
  s = block_sum(s);
  if (threadIdx.x == 0) out[blockIdx.x] = s + b[0];
}

// ---------- host launch ----------
extern "C" void kernel_launch(void* const* d_in, const int* in_sizes, int n_in,
                              void* d_out, int out_size, void* d_ws, size_t ws_size,
                              hipStream_t stream)
{
  (void)in_sizes; (void)n_in; (void)out_size; (void)ws_size;
  const int*   xt   = (const int*)d_in[0];
  const int*   zi   = (const int*)d_in[1];
  const float* pos  = (const float*)d_in[2];
  const float* temb = (const float*)d_in[3];
  const float* Wq   = (const float*)d_in[4];
  const float* Wk   = (const float*)d_in[5];
  const float* Wv   = (const float*)d_in[6];
  const float* g1   = (const float*)d_in[7];
  const float* b1   = (const float*)d_in[8];
  const float* g2   = (const float*)d_in[9];
  const float* b2   = (const float*)d_in[10];
  const float* w1   = (const float*)d_in[11];
  const float* bb1  = (const float*)d_in[12];
  const float* w2   = (const float*)d_in[13];
  const float* bb2  = (const float*)d_in[14];
  const float* row  = (const float*)d_in[15];
  const float* rob  = (const float*)d_in[16];
  float* out = (float*)d_out;

  char* ws = (char*)d_ws;
  const long long SZ_H  = (long long)BATCH * N_TOK * DMODEL * 4;   // 33.5 MB
  const long long SZ_X  = (long long)BATCH * N_TOK * DMODEL * 2;   // 16.8 MB
  float*          H    = (float*)(ws);
  unsigned short* Xbf  = (unsigned short*)(ws + SZ_H);
  char*           big  = ws + SZ_H + SZ_X;                          // 67.1 MB region
  unsigned short* Qbf  = (unsigned short*)(big);
  unsigned short* Kbf  = (unsigned short*)(big + SZ_X);
  unsigned short* Vbf  = (unsigned short*)(big + 2 * SZ_X);
  unsigned short* Vt   = (unsigned short*)(big + 3 * SZ_X);
  unsigned short* Gbf  = (unsigned short*)(big);                    // aliases Q/K/V/Vt (dead by MLP)
  float*          S    = (float*)(big + 4 * SZ_X);                  // 33.5 MB scores
  unsigned short* Pbf  = (unsigned short*)(big + 4 * SZ_X + (long long)BATCH * N_TOK * N_TOK * 4);
  char*           wst  = (char*)Pbf + (long long)BATCH * N_TOK * N_TOK * 2;  // 8.4 MB weight stage
  unsigned short* Wst0 = (unsigned short*)(wst);
  unsigned short* Wst1 = (unsigned short*)(wst + (long long)DMODEL * DMODEL * 2);
  unsigned short* Wst2 = (unsigned short*)(wst + (long long)2 * DMODEL * DMODEL * 2);
  unsigned short* W1t  = (unsigned short*)(wst);   // 8.4 MB, reused sequentially
  unsigned short* W2t  = (unsigned short*)(wst);

  const int MT = BATCH * N_TOK;                 // 8192 rows of activation
  const float inv_sqrt_d = 0.03125f;            // 1/sqrt(1024)

  embed_kernel<<<(BATCH * N_TOK * DMODEL) / 256, 256, 0, stream>>>(xt, zi, pos, temb, H);

  for (int l = 0; l < NLAYER; l++) {
    const long long wq_off = (long long)l * DMODEL * DMODEL;
    // ---- LN1 ----
    layernorm_bf16<<<MT, 256, 0, stream>>>(H, g1 + l * DMODEL, b1 + l * DMODEL, Xbf);
    // ---- Q/K/V projections ----
    transpose_cast_f32_bf16<<<dim3(DMODEL / 32, DMODEL / 32), 256, 0, stream>>>(Wq + wq_off, Wst0, DMODEL, DMODEL);
    transpose_cast_f32_bf16<<<dim3(DMODEL / 32, DMODEL / 32), 256, 0, stream>>>(Wk + wq_off, Wst1, DMODEL, DMODEL);
    transpose_cast_f32_bf16<<<dim3(DMODEL / 32, DMODEL / 32), 256, 0, stream>>>(Wv + wq_off, Wst2, DMODEL, DMODEL);
    gemm_bf16_wmma<0><<<dim3(DMODEL / 256, MT / 128, 1), 256, 0, stream>>>(
        Xbf, Wst0, Qbf, nullptr, MT, DMODEL, DMODEL, 0, 0, 0, 1.f);
    gemm_bf16_wmma<0><<<dim3(DMODEL / 256, MT / 128, 1), 256, 0, stream>>>(
        Xbf, Wst1, Kbf, nullptr, MT, DMODEL, DMODEL, 0, 0, 0, 1.f);
    gemm_bf16_wmma<0><<<dim3(DMODEL / 256, MT / 128, 1), 256, 0, stream>>>(
        Xbf, Wst2, Vbf, nullptr, MT, DMODEL, DMODEL, 0, 0, 0, 1.f);
    // ---- scores = Q K^T / sqrt(D) ----
    gemm_bf16_wmma<1><<<dim3(N_TOK / 256, N_TOK / 128, BATCH), 256, 0, stream>>>(
        Qbf, Kbf, S, nullptr, N_TOK, N_TOK, DMODEL,
        (long long)N_TOK * DMODEL, (long long)N_TOK * DMODEL, (long long)N_TOK * N_TOK, inv_sqrt_d);
    // ---- softmax ----
    softmax_bf16<<<BATCH * N_TOK, 256, 0, stream>>>(S, Pbf);
    // ---- v transpose: [b][t][d] -> [b][d][t] ----
    transpose_bf16_batched<<<dim3(DMODEL / 32, N_TOK / 32, BATCH), 256, 0, stream>>>(Vbf, Vt, N_TOK, DMODEL);
    // ---- H += P V ----
    gemm_bf16_wmma<2><<<dim3(DMODEL / 256, N_TOK / 128, BATCH), 256, 0, stream>>>(
        Pbf, Vt, H, nullptr, N_TOK, DMODEL, N_TOK,
        (long long)N_TOK * N_TOK, (long long)DMODEL * N_TOK, (long long)N_TOK * DMODEL, 1.f);
    // ---- LN2 ----
    layernorm_bf16<<<MT, 256, 0, stream>>>(H, g2 + l * DMODEL, b2 + l * DMODEL, Xbf);
    // ---- MLP up: G = gelu(X W1 + b1) ----
    transpose_cast_f32_bf16<<<dim3(HMLP / 32, DMODEL / 32), 256, 0, stream>>>(
        w1 + (long long)l * DMODEL * HMLP, W1t, DMODEL, HMLP);
    gemm_bf16_wmma<3><<<dim3(HMLP / 256, MT / 128, 1), 256, 0, stream>>>(
        Xbf, W1t, Gbf, bb1 + (long long)l * HMLP, MT, HMLP, DMODEL, 0, 0, 0, 1.f);
    // ---- MLP down: H += G W2 + b2 ----
    transpose_cast_f32_bf16<<<dim3(DMODEL / 32, HMLP / 32), 256, 0, stream>>>(
        w2 + (long long)l * HMLP * DMODEL, W2t, HMLP, DMODEL);
    gemm_bf16_wmma<4><<<dim3(DMODEL / 256, MT / 128, 1), 256, 0, stream>>>(
        Gbf, W2t, H, bb2 + (long long)l * DMODEL, MT, DMODEL, HMLP, 0, 0, 0, 1.f);
  }

  readout_kernel<<<BATCH * N_IMG, 256, 0, stream>>>(H, row, rob, out);
}
